// GATEncoder_30666066493874
// MI455X (gfx1250) — compile-verified, matrix operand order checked by reading, not run on hardware
//
#include <hip/hip_runtime.h>

#define NODES_D   3000
#define HEADS     4
#define CHN       512
#define HC        (HEADS * CHN)   // 2048
#define ZDIM      32
#define SLOPE_ATT 0.2f
#define SLOPE_ACT 0.01f

typedef __attribute__((ext_vector_type(16))) __bf16 bf16x16;
typedef __attribute__((ext_vector_type(8)))  float  f32x8;

union FragU { bf16x16 v; uint4 q[2]; };

__device__ __forceinline__ unsigned int bf16_rne(float f) {
  unsigned int u = __float_as_uint(f);
  return (u + 0x7FFFu + ((u >> 16) & 1u)) >> 16;
}
__device__ __forceinline__ unsigned int pack2(float lo, float hi) {
  return bf16_rne(lo) | (bf16_rne(hi) << 16);
}
__device__ __forceinline__ float leakyf(float x, float s) { return x > 0.0f ? x : x * s; }

// ---------------------------------------------------------------------------
// Elementwise fp32 -> bf16 (for the input features x; done once)
// ---------------------------------------------------------------------------
__global__ void k_f32_to_bf16(const float* __restrict__ in,
                              unsigned short* __restrict__ out, int n) {
  int base = (blockIdx.x * blockDim.x + threadIdx.x) * 8;
  if (base + 8 <= n) {
    const float4* p = (const float4*)(in + base);
    float4 x0 = p[0], x1 = p[1];
    uint4 q;
    q.x = pack2(x0.x, x0.y); q.y = pack2(x0.z, x0.w);
    q.z = pack2(x1.x, x1.y); q.w = pack2(x1.z, x1.w);
    *(uint4*)(out + base) = q;
  } else {
    for (int j = base; j < n; ++j) out[j] = (unsigned short)bf16_rne(in[j]);
  }
}

// ---------------------------------------------------------------------------
// Weight transpose + fp32 -> bf16:  W[K][Nt] (fp32) -> Wt[Nt][K] (bf16)
// ---------------------------------------------------------------------------
__global__ void k_transpose_bf16(const float* __restrict__ W,
                                 unsigned short* __restrict__ Wt,
                                 int K, int Nt) {
  __shared__ float tile[32][33];
  int tx = threadIdx.x, ty = threadIdx.y;          // 32 x 8
  int nBase = blockIdx.x * 32, kBase = blockIdx.y * 32;
  #pragma unroll
  for (int j = 0; j < 32; j += 8) {
    int k = kBase + ty + j, c = nBase + tx;
    if (k < K && c < Nt) tile[ty + j][tx] = W[(size_t)k * Nt + c];
  }
  __syncthreads();
  #pragma unroll
  for (int j = 0; j < 32; j += 8) {
    int c = nBase + ty + j, k = kBase + tx;
    if (c < Nt && k < K)
      Wt[(size_t)c * K + k] = (unsigned short)bf16_rne(tile[tx][ty + j]);
  }
}

// ---------------------------------------------------------------------------
// Tiled bf16 WMMA GEMM:  C[M][Nt](fp32) = A[M][K](bf16) * Bt[Nt][K](bf16)^T
// Block tile 256x128, K-step 32. 8 waves as 4(m) x 2(n); each wave owns a
// 64x64 patch = 4x4 WMMA 16x16x32 bf16 fragments (16 WMMA / wave / k-step),
// fp32 accumulation, double-buffered fragment-ready LDS:
//   frag slot = [frag][lane][16 bf16] -> two ds_*_b128 per lane.
// ---------------------------------------------------------------------------
__global__ __launch_bounds__(256)
void k_gemm_bf16(const unsigned short* __restrict__ A,
                 const unsigned short* __restrict__ Bt,
                 float* __restrict__ C, int M, int Nt, int K) {
  __shared__ uint4 As[2][16][32][2];   // 32 KB
  __shared__ uint4 Bs[2][8][32][2];    // 16 KB

  const int tid  = threadIdx.x;
  const int wave = tid >> 5, lane = tid & 31;
  const int wm = wave & 3, wn = wave >> 2;
  const int bM = blockIdx.x * 256, bN = blockIdx.y * 128;

  // staging roles: thread t fills A-frag slots (t>>5) and (t>>5)+8, B-frag slot (t>>5)
  const int cl    = tid & 31;
  const int kbase = (cl >> 4) * 8;
  const int fr    = tid >> 5;                      // 0..7
  const int arow0 = min(bM + fr * 16       + (cl & 15), M - 1);
  const int arow1 = min(bM + (fr + 8) * 16 + (cl & 15), M - 1);
  const int brow  = min(bN + fr * 16       + (cl & 15), Nt - 1);
  const unsigned short* ap0 = A  + (size_t)arow0 * K;
  const unsigned short* ap1 = A  + (size_t)arow1 * K;
  const unsigned short* bp  = Bt + (size_t)brow  * K;

  f32x8 acc[4][4];
  #pragma unroll
  for (int i = 0; i < 4; ++i)
    #pragma unroll
    for (int j = 0; j < 4; ++j)
      acc[i][j] = (f32x8){0.f,0.f,0.f,0.f,0.f,0.f,0.f,0.f};

  const int steps = (K + 31) / 32;

  auto stage = [&](const unsigned short* p, int k0) -> uint4 {
    if (k0 + 8 <= K) return *(const uint4*)(p + k0);
    unsigned int v[8];
    #pragma unroll
    for (int j = 0; j < 8; ++j) { int k = k0 + j; v[j] = (k < K) ? (unsigned int)p[k] : 0u; }
    uint4 q;
    q.x = v[0] | (v[1] << 16); q.y = v[2] | (v[3] << 16);
    q.z = v[4] | (v[5] << 16); q.w = v[6] | (v[7] << 16);
    return q;
  };

  auto loadTile = [&](int s, int buf) {
    const int kk = s * 32;
    if (kk + 64 + kbase < K) {              // prefetch k-step s+2 into caches
      __builtin_prefetch(ap0 + kk + 64 + kbase, 0, 1);
      __builtin_prefetch(bp  + kk + 64 + kbase, 0, 1);
    }
    #pragma unroll
    for (int run = 0; run < 2; ++run) {
      const int k0 = kk + kbase + run * 16;
      As[buf][fr][cl][run]     = stage(ap0, k0);
      As[buf][fr + 8][cl][run] = stage(ap1, k0);
      Bs[buf][fr][cl][run]     = stage(bp,  k0);
    }
  };

  loadTile(0, 0);
  __syncthreads();

  for (int s = 0; s < steps; ++s) {
    const int cur = s & 1;
    if (s + 1 < steps) loadTile(s + 1, cur ^ 1);

    FragU a[4];
    #pragma unroll
    for (int i = 0; i < 4; ++i) {
      a[i].q[0] = As[cur][4 * wm + i][lane][0];
      a[i].q[1] = As[cur][4 * wm + i][lane][1];
    }
    #pragma unroll
    for (int j = 0; j < 4; ++j) {
      FragU b;
      b.q[0] = Bs[cur][4 * wn + j][lane][0];
      b.q[1] = Bs[cur][4 * wn + j][lane][1];
      #pragma unroll
      for (int i = 0; i < 4; ++i)
        acc[i][j] = __builtin_amdgcn_wmma_f32_16x16x32_bf16(
            false, a[i].v, false, b.v, (short)0, acc[i][j], false, false);
    }
    __syncthreads();
  }

  // epilogue: C/D layout — VGPR e: row = e + 8*(lane>=16), col = lane&15
  const int rowadd = 8 * (lane >> 4);
  const int colin  = lane & 15;
  #pragma unroll
  for (int j = 0; j < 4; ++j) {
    const int c = bN + (4 * wn + j) * 16 + colin;
    if (c < Nt) {
      #pragma unroll
      for (int i = 0; i < 4; ++i) {
        const int r0 = bM + (4 * wm + i) * 16 + rowadd;
        #pragma unroll
        for (int e = 0; e < 8; ++e) {
          const int r = r0 + e;
          if (r < M) C[(size_t)r * Nt + c] = acc[i][j][e];
        }
      }
    }
  }
}

// ---------------------------------------------------------------------------
// Attention logits: alpha_s[n,h] = <h[n,h,:], a_src[h,:]>, same for alpha_d.
// One wave per (node, head), wave-reduce over C=512.
// ---------------------------------------------------------------------------
__global__ void k_alpha(const float* __restrict__ h,
                        const float* __restrict__ a_s,
                        const float* __restrict__ a_d,
                        float* __restrict__ out_s, float* __restrict__ out_d,
                        int nTot) {
  int gw = (int)((blockIdx.x * blockDim.x + threadIdx.x) >> 5);
  int lane = threadIdx.x & 31;
  if (gw >= nTot) return;                       // uniform per wave
  int node = gw / HEADS, head = gw % HEADS;
  const float* hp  = h   + (size_t)node * HC + head * CHN;
  const float* asp = a_s + head * CHN;
  const float* adp = a_d + head * CHN;
  float ss = 0.f, sd = 0.f;
  for (int c = lane; c < CHN; c += 32) {
    float v = hp[c];
    ss += v * asp[c];
    sd += v * adp[c];
  }
  for (int off = 16; off; off >>= 1) {
    ss += __shfl_down(ss, off);
    sd += __shfl_down(sd, off);
  }
  if (lane == 0) { out_s[gw] = ss; out_d[gw] = sd; }
}

// ---------------------------------------------------------------------------
// CSR build (by dst, with self-loops)
// ---------------------------------------------------------------------------
__global__ void k_deg_init(int* __restrict__ deg, int n) {
  int t = blockIdx.x * blockDim.x + threadIdx.x;
  if (t < n) deg[t] = 1;                        // self-loop
}
__global__ void k_deg_count(const int* __restrict__ dst, int* __restrict__ deg, int e) {
  int t = blockIdx.x * blockDim.x + threadIdx.x;
  if (t < e) atomicAdd(&deg[dst[t]], 1);
}
__global__ __launch_bounds__(1024)
void k_scan(const int* __restrict__ deg, int* __restrict__ rowptr,
            int* __restrict__ cursor, int n) {
  __shared__ int sums[1024];
  int t = threadIdx.x;
  int ch = (n + 1023) >> 10;
  int base = t * ch;
  int local = 0;
  for (int j = 0; j < ch; ++j) { int i = base + j; if (i < n) local += deg[i]; }
  sums[t] = local;
  __syncthreads();
  for (int off = 1; off < 1024; off <<= 1) {
    int v = (t >= off) ? sums[t - off] : 0;
    __syncthreads();
    sums[t] += v;
    __syncthreads();
  }
  int run = (t == 0) ? 0 : sums[t - 1];
  for (int j = 0; j < ch; ++j) {
    int i = base + j;
    if (i < n) { rowptr[i] = run; cursor[i] = run; run += deg[i]; }
  }
  if (t == 1023) rowptr[n] = sums[1023];
}
__global__ void k_scatter(const int* __restrict__ src, const int* __restrict__ dst,
                          int* __restrict__ cursor, int* __restrict__ csr_src,
                          int e, int n) {
  int t = blockIdx.x * blockDim.x + threadIdx.x;
  if (t < e) {
    int pos = atomicAdd(&cursor[dst[t]], 1);
    csr_src[pos] = src[t];
  } else if (t < e + n) {
    int i = t - e;
    int pos = atomicAdd(&cursor[i], 1);
    csr_src[pos] = i;                           // self-loop
  }
}

// ---------------------------------------------------------------------------
// Per-node softmax attention + aggregation. One 256-thread block per node.
// h[src] rows gathered from global (whole h tensor resides in 192MB L2).
// Output is written in bf16 (it feeds only the next bf16 GEMM).
// concatMode=1: out[n][HC] (+bias, optional leaky); 0: head-mean out[n][CHN].
// ---------------------------------------------------------------------------
__global__ __launch_bounds__(256)
void k_agg(const float* __restrict__ h, const float* __restrict__ als,
           const float* __restrict__ ald, const int* __restrict__ rowptr,
           const int* __restrict__ csr_src, const float* __restrict__ bias,
           unsigned short* __restrict__ out, int concatMode, int actMode) {
  const int i = blockIdx.x;
  const int tid = threadIdx.x;
  const int beg = rowptr[i], end = rowptr[i + 1];

  __shared__ float s_m[HEADS], s_d[HEADS];
  __shared__ int   s_src[256];
  __shared__ float s_red[256 * 8];

  if (tid < 32) {
    const int lane = tid;
    for (int hh = 0; hh < HEADS; ++hh) {
      const float adv = ald[i * HEADS + hh];
      float m = -1e30f;
      for (int e = beg + lane; e < end; e += 32) {
        float v = leakyf(als[csr_src[e] * HEADS + hh] + adv, SLOPE_ATT);
        m = fmaxf(m, v);
      }
      for (int off = 16; off; off >>= 1) m = fmaxf(m, __shfl_down(m, off));
      m = __shfl(m, 0);
      float sum = 0.f;
      for (int e = beg + lane; e < end; e += 32) {
        float v = leakyf(als[csr_src[e] * HEADS + hh] + adv, SLOPE_ATT);
        sum += __expf(v - m);
      }
      for (int off = 16; off; off >>= 1) sum += __shfl_down(sum, off);
      if (lane == 0) { s_m[hh] = m; s_d[hh] = sum; }
    }
  }
  __syncthreads();

  const int head = tid >> 6;              // 64 threads per head
  const int c0   = (tid & 63) << 3;       // 8 channels per thread
  const float adv = ald[i * HEADS + head];
  const float m   = s_m[head];
  const float inv = 1.0f / (s_d[head] + 1e-16f);

  float acc[8] = {0.f, 0.f, 0.f, 0.f, 0.f, 0.f, 0.f, 0.f};

  for (int cbeg = beg; cbeg < end; cbeg += 256) {
    const int cnt = min(256, end - cbeg);
    __syncthreads();
    if (tid < cnt) s_src[tid] = csr_src[cbeg + tid];
    __syncthreads();
    for (int e = 0; e < cnt; ++e) {
      const int s = s_src[e];
      const float v = leakyf(als[s * HEADS + head] + adv, SLOPE_ATT);
      const float w = __expf(v - m) * inv;
      const float4* hp = (const float4*)(h + (size_t)s * HC + head * CHN + c0);
      float4 p0 = hp[0], p1 = hp[1];
      acc[0] += p0.x * w; acc[1] += p0.y * w; acc[2] += p0.z * w; acc[3] += p0.w * w;
      acc[4] += p1.x * w; acc[5] += p1.y * w; acc[6] += p1.z * w; acc[7] += p1.w * w;
    }
  }

  if (concatMode) {
    const int col = head * CHN + c0;
    float o[8];
    #pragma unroll
    for (int j = 0; j < 8; ++j) {
      o[j] = acc[j] + bias[col + j];
      if (actMode) o[j] = leakyf(o[j], SLOPE_ACT);
    }
    uint4 q;
    q.x = pack2(o[0], o[1]); q.y = pack2(o[2], o[3]);
    q.z = pack2(o[4], o[5]); q.w = pack2(o[6], o[7]);
    *(uint4*)(out + (size_t)i * HC + col) = q;
  } else {
    #pragma unroll
    for (int j = 0; j < 8; ++j) s_red[tid * 8 + j] = acc[j];
    __syncthreads();
    if (tid < 64) {
      float o[8];
      #pragma unroll
      for (int j = 0; j < 8; ++j) {
        const int idx = tid * 8 + j;
        o[j] = (s_red[idx] + s_red[(tid + 64) * 8 + j] +
                s_red[(tid + 128) * 8 + j] + s_red[(tid + 192) * 8 + j]) * 0.25f
               + bias[idx];
      }
      uint4 q;
      q.x = pack2(o[0], o[1]); q.y = pack2(o[2], o[3]);
      q.z = pack2(o[4], o[5]); q.w = pack2(o[6], o[7]);
      *(uint4*)(out + (size_t)i * CHN + tid * 8) = q;
    }
  }
}

// ---------------------------------------------------------------------------
// Final head: z[N][2*ZD] -> mu = z[:, :ZD]+bfc, sigma2 = exp(z[:, ZD:]+bfc)
// ---------------------------------------------------------------------------
__global__ void k_head(const float* __restrict__ z, const float* __restrict__ bfc,
                       float* __restrict__ out, int n) {
  int t = blockIdx.x * blockDim.x + threadIdx.x;
  if (t >= n * ZDIM) return;
  int i = t / ZDIM, j = t % ZDIM;
  float mu = z[(size_t)i * (2 * ZDIM) + j] + bfc[j];
  float lv = z[(size_t)i * (2 * ZDIM) + ZDIM + j] + bfc[ZDIM + j];
  out[t] = mu;
  out[(size_t)n * ZDIM + t] = __expf(lv);
}

// ---------------------------------------------------------------------------
// Host launcher
// ---------------------------------------------------------------------------
extern "C" void kernel_launch(void* const* d_in, const int* in_sizes, int n_in,
                              void* d_out, int out_size, void* d_ws, size_t ws_size,
                              hipStream_t stream) {
  const float* x   = (const float*)d_in[0];
  const int*   ei  = (const int*)  d_in[1];
  const float* W1  = (const float*)d_in[2];
  const float* as1 = (const float*)d_in[3];
  const float* ad1 = (const float*)d_in[4];
  const float* b1  = (const float*)d_in[5];
  const float* W2  = (const float*)d_in[6];
  const float* as2 = (const float*)d_in[7];
  const float* ad2 = (const float*)d_in[8];
  const float* b2  = (const float*)d_in[9];
  const float* W3  = (const float*)d_in[10];
  const float* as3 = (const float*)d_in[11];
  const float* ad3 = (const float*)d_in[12];
  const float* b3  = (const float*)d_in[13];
  const float* Wfc = (const float*)d_in[14];
  const float* bfc = (const float*)d_in[15];

  const int nN = in_sizes[0] / NODES_D;   // 10000
  const int nE = in_sizes[1] / 2;         // 60000
  const int* srcE = ei;
  const int* dstE = ei + nE;

  // workspace carving (256B aligned)
  char* ws = (char*)d_ws;
  size_t off = 0;
  auto carve = [&](size_t bytes) -> void* {
    void* p = ws + off;
    off += (bytes + 255) & ~(size_t)255;
    return p;
  };
  unsigned short* xbf    = (unsigned short*)carve((size_t)nN * NODES_D * sizeof(unsigned short));
  float*          bufH   = (float*)         carve((size_t)nN * HC * sizeof(float));
  unsigned short* bufOb  = (unsigned short*)carve((size_t)nN * HC * sizeof(unsigned short));
  unsigned short* Wt     = (unsigned short*)carve((size_t)NODES_D * HC * sizeof(unsigned short));
  float*          aS     = (float*)         carve((size_t)nN * HEADS * sizeof(float));
  float*          aD     = (float*)         carve((size_t)nN * HEADS * sizeof(float));
  int*            deg    = (int*)           carve((size_t)nN * sizeof(int));
  int*            rowptr = (int*)           carve((size_t)(nN + 1) * sizeof(int));
  int*            cursor = (int*)           carve((size_t)nN * sizeof(int));
  int*            csr    = (int*)           carve((size_t)(nE + nN) * sizeof(int));

  const dim3 tB(32, 8);

  // ---- CSR build (once; shared by all 3 layers) ----
  k_deg_init <<<(nN + 255) / 256, 256, 0, stream>>>(deg, nN);
  k_deg_count<<<(nE + 255) / 256, 256, 0, stream>>>(dstE, deg, nE);
  k_scan     <<<1, 1024, 0, stream>>>(deg, rowptr, cursor, nN);
  k_scatter  <<<(nE + nN + 255) / 256, 256, 0, stream>>>(srcE, dstE, cursor, csr, nE, nN);

  // ---- x -> bf16 (once) ----
  {
    int n = nN * NODES_D;
    k_f32_to_bf16<<<(n / 8 + 255) / 256, 256, 0, stream>>>(x, xbf, n);
  }

  const int gM = (nN + 255) / 256;
  const int alphaBlocks = (nN * HEADS * 32 + 255) / 256;

  // ---- Layer 1: K=3000, Nt=2048 ----
  k_transpose_bf16<<<dim3((HC + 31) / 32, (NODES_D + 31) / 32), tB, 0, stream>>>(W1, Wt, NODES_D, HC);
  k_gemm_bf16<<<dim3(gM, HC / 128), 256, 0, stream>>>(xbf, Wt, bufH, nN, HC, NODES_D);
  k_alpha<<<alphaBlocks, 256, 0, stream>>>(bufH, as1, ad1, aS, aD, nN * HEADS);
  k_agg<<<nN, 256, 0, stream>>>(bufH, aS, aD, rowptr, csr, b1, bufOb, 1, 1);

  // ---- Layer 2: K=2048, Nt=2048 ----
  k_transpose_bf16<<<dim3((HC + 31) / 32, (HC + 31) / 32), tB, 0, stream>>>(W2, Wt, HC, HC);
  k_gemm_bf16<<<dim3(gM, HC / 128), 256, 0, stream>>>(bufOb, Wt, bufH, nN, HC, HC);
  k_alpha<<<alphaBlocks, 256, 0, stream>>>(bufH, as2, ad2, aS, aD, nN * HEADS);
  k_agg<<<nN, 256, 0, stream>>>(bufH, aS, aD, rowptr, csr, b2, bufOb, 1, 1);

  // ---- Layer 3: K=2048, Nt=2048, head-mean output (N x 512, bf16) ----
  k_transpose_bf16<<<dim3((HC + 31) / 32, (HC + 31) / 32), tB, 0, stream>>>(W3, Wt, HC, HC);
  k_gemm_bf16<<<dim3(gM, HC / 128), 256, 0, stream>>>(bufOb, Wt, bufH, nN, HC, HC);
  k_alpha<<<alphaBlocks, 256, 0, stream>>>(bufH, as3, ad3, aS, aD, nN * HEADS);
  k_agg<<<nN, 256, 0, stream>>>(bufH, aS, aD, rowptr, csr, b3, bufOb, 0, 0);

  // ---- FC head: (N x 512, bf16) @ (512 x 64) ----
  k_transpose_bf16<<<dim3((2 * ZDIM + 31) / 32, (CHN + 31) / 32), tB, 0, stream>>>(Wfc, Wt, CHN, 2 * ZDIM);
  k_gemm_bf16<<<dim3(gM, 1), 256, 0, stream>>>(bufOb, Wt, bufH, nN, 2 * ZDIM, CHN);
  k_head<<<(nN * ZDIM + 255) / 256, 256, 0, stream>>>(bufH, bfc, (float*)d_out, nN);
}
